// CLIP_SVPR_ReID_14525579395180
// MI455X (gfx1250) — compile-verified, matrix operand order checked by reading, not run on hardware
//
#include <hip/hip_runtime.h>
#include <hip/hip_bf16.h>

// ---------------------------------------------------------------------------
// Problem constants (match reference)
// ---------------------------------------------------------------------------
#define Bn 256
#define Nn 256
#define Cn 768
#define An 10
#define Hn 8
#define Dn 96      // C / H
#define En 4
#define TOPK_ATTR 7
#define MOE_TOPK 3

typedef __attribute__((ext_vector_type(16))) __bf16        v16bf;
typedef __attribute__((ext_vector_type(8)))  float         v8f;
typedef __attribute__((ext_vector_type(4)))  unsigned int  v4u;
typedef __attribute__((ext_vector_type(8)))  int           v8i;
typedef __attribute__((ext_vector_type(4)))  int           v4i;

__device__ __forceinline__ float wred(float v) {
#pragma unroll
    for (int off = 16; off > 0; off >>= 1) v += __shfl_xor(v, off, 32);
    return v;
}
__device__ __forceinline__ float wredmax(float v) {
#pragma unroll
    for (int off = 16; off > 0; off >>= 1) v = fmaxf(v, __shfl_xor(v, off, 32));
    return v;
}

// ---------------------------------------------------------------------------
// bf16 fragment loader per CDNA5 ISA 16-bit A/B layout:
//   lane L: row = L%16, kbase = (L>=16)*8;
//   f[j] (j<4) packs K = kbase+2j, +1 ; f[4+j] packs K = 16+kbase+2j, +1
// ---------------------------------------------------------------------------
__device__ __forceinline__ v16bf load_frag(const float* __restrict__ row,
                                           int kc, int kb) {
    v16bf f;
#pragma unroll
    for (int j = 0; j < 4; ++j) {
        const int k0 = kc + kb + 2 * j;
        const int k1 = k0 + 16;
        f[2 * j]         = (__bf16)row[k0];
        f[2 * j + 1]     = (__bf16)row[k0 + 1];
        f[8 + 2 * j]     = (__bf16)row[k1];
        f[8 + 2 * j + 1] = (__bf16)row[k1 + 1];
    }
    return f;
}

// ---------------------------------------------------------------------------
// Generic WMMA GEMM:  C[m,n] (op)= sum_k A[m,k] * Bw[n,k]  (+ bias[n])
// A: [M,K] row-major (lda), Bw: [N,K] row-major (torch Linear weight, ldb).
// One wave per 32x32 output tile (2x2 of 16x16): each A/B fragment is reused
// by two v_wmma_f32_16x16x32_bf16, halving per-WMMA load traffic.
// K multiple of 32 (768, 96, 256 all are); M,N multiples of 32.
// ---------------------------------------------------------------------------
template <bool ACC>
__global__ void wmma_gemm_bt(const float* __restrict__ A, int lda,
                             const float* __restrict__ Bw, int ldb,
                             const float* __restrict__ bias,
                             float* __restrict__ Cmat, int ldc, int K) {
    const int lane = threadIdx.x & 31;
    const int r    = lane & 15;
    const int hi   = lane >> 4;
    const int kb   = hi * 8;
    const int m0   = blockIdx.x * 32;
    const int n0   = blockIdx.y * 32;

    const float* a0 = A  + (size_t)(m0 + r) * lda;
    const float* a1 = a0 + (size_t)16 * lda;
    const float* b0 = Bw + (size_t)(n0 + r) * ldb;
    const float* b1 = b0 + (size_t)16 * ldb;

    v8f c00, c01, c10, c11;
    if (ACC) {
#pragma unroll
        for (int j = 0; j < 8; ++j) {
            const size_t row0 = (size_t)(m0 + hi * 8 + j) * ldc;
            const size_t row1 = row0 + (size_t)16 * ldc;
            c00[j] = Cmat[row0 + n0 + r];
            c01[j] = Cmat[row0 + n0 + 16 + r];
            c10[j] = Cmat[row1 + n0 + r];
            c11[j] = Cmat[row1 + n0 + 16 + r];
        }
    } else if (bias) {
        const float bv0 = bias[n0 + r];
        const float bv1 = bias[n0 + 16 + r];
#pragma unroll
        for (int j = 0; j < 8; ++j) { c00[j] = bv0; c01[j] = bv1; c10[j] = bv0; c11[j] = bv1; }
    } else {
#pragma unroll
        for (int j = 0; j < 8; ++j) { c00[j] = 0.f; c01[j] = 0.f; c10[j] = 0.f; c11[j] = 0.f; }
    }

    for (int kc = 0; kc < K; kc += 32) {
        const v16bf fa0 = load_frag(a0, kc, kb);
        const v16bf fa1 = load_frag(a1, kc, kb);
        const v16bf fb0 = load_frag(b0, kc, kb);
        const v16bf fb1 = load_frag(b1, kc, kb);
        c00 = __builtin_amdgcn_wmma_f32_16x16x32_bf16(false, fa0, false, fb0, (short)0, c00, false, false);
        c01 = __builtin_amdgcn_wmma_f32_16x16x32_bf16(false, fa0, false, fb1, (short)0, c01, false, false);
        c10 = __builtin_amdgcn_wmma_f32_16x16x32_bf16(false, fa1, false, fb0, (short)0, c10, false, false);
        c11 = __builtin_amdgcn_wmma_f32_16x16x32_bf16(false, fa1, false, fb1, (short)0, c11, false, false);
    }

#pragma unroll
    for (int j = 0; j < 8; ++j) {
        const size_t row0 = (size_t)(m0 + hi * 8 + j) * ldc;
        const size_t row1 = row0 + (size_t)16 * ldc;
        Cmat[row0 + n0 + r]      = c00[j];
        Cmat[row0 + n0 + 16 + r] = c01[j];
        Cmat[row1 + n0 + r]      = c10[j];
        Cmat[row1 + n0 + 16 + r] = c11[j];
    }
}

// ---------------------------------------------------------------------------
// q2[b,h,:] = q[b, h*96 .. h*96+95] @ Wk[h*96 .. , :]   (fold q through Wk)
// grid = B, block = 256. Wk is L2-resident (2.25 MB).
// ---------------------------------------------------------------------------
__global__ void qhead_kernel(const float* __restrict__ q,
                             const float* __restrict__ Wk,
                             float* __restrict__ q2) {
    __shared__ float qs[Cn];
    const int b = blockIdx.x, tid = threadIdx.x;
    for (int i = tid; i < Cn; i += 256) qs[i] = q[(size_t)b * Cn + i];
    __syncthreads();
#pragma unroll 1
    for (int h = 0; h < Hn; ++h) {
#pragma unroll
        for (int cc = 0; cc < 3; ++cc) {
            const int c = tid + cc * 256;
            float acc = 0.0f;
            for (int d = 0; d < Dn; ++d)
                acc = fmaf(qs[h * Dn + d], Wk[(size_t)(h * Dn + d) * Cn + c], acc);
            q2[((size_t)b * Hn + h) * Cn + c] = acc;
        }
    }
}

// ---------------------------------------------------------------------------
// Streaming pass 1: scores + softmax.
// score(b,h,n) = scale * ( q2[b,h,:] . x[b,n,:]  +  q[b,h*96:] . bk[h*96:] )
// att = softmax over n.  grid = B, block = 256 (8 waves).
// ---------------------------------------------------------------------------
__global__ void scores_softmax_kernel(const float* __restrict__ q2,
                                      const float* __restrict__ x,
                                      const float* __restrict__ q,
                                      const float* __restrict__ bk,
                                      float* __restrict__ att) {
    __shared__ float q2s[Hn * Cn];   // 24 KB
    __shared__ float sc[Hn * Nn];    // 8 KB
    __shared__ float qbk[Hn];
    const int b = blockIdx.x, tid = threadIdx.x;
    const int wave = tid >> 5, lane = tid & 31;

    for (int i = tid; i < Hn * Cn; i += 256) q2s[i] = q2[(size_t)b * Hn * Cn + i];
    if (tid < Hn) {
        float s = 0.0f;
        for (int d = 0; d < Dn; ++d)
            s = fmaf(q[(size_t)b * Cn + tid * Dn + d], bk[tid * Dn + d], s);
        qbk[tid] = s;
    }
    __syncthreads();

    const float scale = rsqrtf((float)Dn);
    for (int n = wave; n < Nn; n += 8) {
        const float* xr = x + ((size_t)b * Nn + n) * Cn;
        if (n + 8 < Nn)  // global_prefetch_b8 the next row this wave will use
            __builtin_prefetch(xr + (size_t)8 * Cn + lane * 16, 0, 1);
        float p[Hn];
#pragma unroll
        for (int h = 0; h < Hn; ++h) p[h] = 0.0f;
        for (int c = lane; c < Cn; c += 32) {
            const float xv = xr[c];
#pragma unroll
            for (int h = 0; h < Hn; ++h) p[h] = fmaf(q2s[h * Cn + c], xv, p[h]);
        }
#pragma unroll
        for (int h = 0; h < Hn; ++h) p[h] = wred(p[h]);
        if (lane == 0) {
#pragma unroll
            for (int h = 0; h < Hn; ++h)
                sc[h * Nn + n] = (p[h] + qbk[h]) * scale;
        }
    }
    __syncthreads();

    // softmax: wave w owns head h = w
    const int h = wave;
    float mx = -INFINITY;
    for (int n = lane; n < Nn; n += 32) mx = fmaxf(mx, sc[h * Nn + n]);
    mx = wredmax(mx);
    float e[Nn / 32];
    float sum = 0.0f;
#pragma unroll
    for (int i = 0; i < Nn / 32; ++i) {
        e[i] = __expf(sc[h * Nn + lane + 32 * i] - mx);
        sum += e[i];
    }
    sum = wred(sum);
    const float inv = 1.0f / sum;
#pragma unroll
    for (int i = 0; i < Nn / 32; ++i)
        att[((size_t)b * Hn + h) * Nn + lane + 32 * i] = e[i] * inv;
}

// ---------------------------------------------------------------------------
// Streaming pass 2: xbar[b,h,c] = sum_n att[b,h,n] * x[b,n,c]
// grid = B, block = 256; 24 accumulators per thread.
// TDM path: double-buffered TENSOR_LOAD_TO_LDS of 8x768 f32 tiles, wave 0
// issues the DMA + s_wait_tensorcnt, workgroup barrier hands tile to all waves.
// ---------------------------------------------------------------------------
#define XROWS   8
#define XCHUNKS (Nn / XROWS)

#if __has_builtin(__builtin_amdgcn_tensor_load_to_lds) && __has_builtin(__builtin_amdgcn_s_wait_tensorcnt)
#define HAVE_TDM 1
#else
#define HAVE_TDM 0
#endif

#if HAVE_TDM
// Issue one TDM load of XROWS x Cn f32 from gaddr into LDS byte-offset lds_off.
// D# group0: count=1, lds_addr, global_addr, type=2 ("image").
// D# group1: data_size=2 (4B), tensor_dim0=Cn, tensor_dim1=XROWS,
//            tile_dim0=Cn, tile_dim1=XROWS, tensor_dim0_stride=Cn.
__device__ __forceinline__ void tdm_load_tile(const float* gaddr, unsigned lds_off) {
    const unsigned long long ga = (unsigned long long)(uintptr_t)gaddr;
    v4u g0 = { 1u,                                   // count = 1 (valid D#)
               lds_off,                              // lds_addr [63:32]
               (unsigned)ga,                         // global_addr lo
               (unsigned)((ga >> 32) & 0x1FFFFFFu) | 0x80000000u };  // ga hi | type=2
    v8i g1 = { 0x20000,                              // data_size = 2 (4 bytes)
               (int)(Cn << 16),                      // tensor_dim0[15:0] @ [63:48]
               (int)(XROWS << 16),                   // tensor_dim1[15:0] @ [95:80]
               (int)(Cn << 16),                      // tile_dim0 @ [127:112]
               XROWS,                                // tile_dim1 @ [143:128]
               Cn,                                   // tensor_dim0_stride @ [207:160]
               0, 0 };
    v4i z4 = { 0, 0, 0, 0 };
#if __clang_major__ >= 23
    v8i z8 = { 0, 0, 0, 0, 0, 0, 0, 0 };
    __builtin_amdgcn_tensor_load_to_lds(g0, g1, z4, z4, z8, 0);
#else
    __builtin_amdgcn_tensor_load_to_lds(g0, g1, z4, z4, 0);
#endif
}
#endif

__global__ void xbar_kernel(const float* __restrict__ att,
                            const float* __restrict__ x,
                            float* __restrict__ xbar) {
#if HAVE_TDM
    __shared__ float smem[2 * XROWS * Cn + Hn * Nn];   // 48 KB tiles + 8 KB att
    float* as_ = smem + 2 * XROWS * Cn;
#else
    __shared__ float as_[Hn * Nn];
#endif
    const int b = blockIdx.x, tid = threadIdx.x;

    for (int i = tid; i < Hn * Nn; i += 256) as_[i] = att[(size_t)b * Hn * Nn + i];

    float acc[3][Hn];
#pragma unroll
    for (int cc = 0; cc < 3; ++cc)
#pragma unroll
        for (int h = 0; h < Hn; ++h) acc[cc][h] = 0.0f;

#if HAVE_TDM
    const float* xb_base = x + (size_t)b * Nn * Cn;
    const unsigned lds_base   = (unsigned)(uintptr_t)&smem[0];
    const unsigned lds_stride = (unsigned)(XROWS * Cn * sizeof(float));
    if (tid < 32)   // wave 0 drives the DMA pipeline
        tdm_load_tile(xb_base, lds_base);

    for (int chunk = 0; chunk < XCHUNKS; ++chunk) {
        if (tid < 32) {
            if (chunk + 1 < XCHUNKS) {
                tdm_load_tile(xb_base + (size_t)(chunk + 1) * XROWS * Cn,
                              lds_base + (unsigned)((chunk + 1) & 1) * lds_stride);
                __builtin_amdgcn_s_wait_tensorcnt(1);  // oldest (current) tile done
            } else {
                __builtin_amdgcn_s_wait_tensorcnt(0);
            }
        }
        __syncthreads();                               // tile visible to all waves
        const float* xt_c = smem + (size_t)(chunk & 1) * (XROWS * Cn);
        const int nbase = chunk * XROWS;
#pragma unroll 1
        for (int nn = 0; nn < XROWS; ++nn) {
#pragma unroll
            for (int cc = 0; cc < 3; ++cc) {
                const float xv = xt_c[nn * Cn + tid + cc * 256];
#pragma unroll
                for (int h = 0; h < Hn; ++h)
                    acc[cc][h] = fmaf(as_[h * Nn + nbase + nn], xv, acc[cc][h]);
            }
        }
        __syncthreads();                               // done reading before overwrite
    }
#else
    __syncthreads();
    for (int n = 0; n < Nn; ++n) {
        const float* xr = x + ((size_t)b * Nn + n) * Cn;
#pragma unroll
        for (int cc = 0; cc < 3; ++cc) {
            const float xv = xr[tid + cc * 256];
#pragma unroll
            for (int h = 0; h < Hn; ++h)
                acc[cc][h] = fmaf(as_[h * Nn + n], xv, acc[cc][h]);
        }
    }
#endif

#pragma unroll
    for (int cc = 0; cc < 3; ++cc)
#pragma unroll
        for (int h = 0; h < Hn; ++h)
            xbar[((size_t)b * Hn + h) * Cn + tid + cc * 256] = acc[cc][h];
}

// ---------------------------------------------------------------------------
// Fused gate / router / top-k combine. Emits per-expert folded vectors
// z[b,e,:] = sum_k wts_k * w[b,a_k,e] * attr_in[b,a_k,:]
// and d_out init = sum_e ce[e] * exp_b[e,:].   grid = B, block = 256.
// ---------------------------------------------------------------------------
__global__ void moe_router_kernel(const float* __restrict__ moe_in,
                                  const float* __restrict__ vcls,
                                  const float* __restrict__ prompt,
                                  const float* __restrict__ gate_w,
                                  const float* __restrict__ gate_b,
                                  const float* __restrict__ r1_w,
                                  const float* __restrict__ r1_b,
                                  const float* __restrict__ r2_w,
                                  const float* __restrict__ r2_b,
                                  const float* __restrict__ exp_b,
                                  float* __restrict__ z,
                                  float* __restrict__ out_init) {
    __shared__ float attr[An * Cn];  // 30 KB
    __shared__ float g[An * En];
    __shared__ float hval[An];
    __shared__ float w[An * En];
    __shared__ float wts[TOPK_ATTR];
    __shared__ int   ti[TOPK_ATTR];
    __shared__ float ce[En];
    __shared__ float r2stats[2];

    const int b = blockIdx.x, tid = threadIdx.x;
    const int wave = tid >> 5, lane = tid & 31;

    // attr_in[a,c] = moe_in[b,c] + visual_cls[b,c] + prompt[a,c]
#pragma unroll
    for (int cc = 0; cc < 3; ++cc) {
        const int c = tid + cc * 256;
        const float base = moe_in[(size_t)b * Cn + c] + vcls[(size_t)b * Cn + c];
        for (int a = 0; a < An; ++a)
            attr[a * Cn + c] = base + prompt[a * Cn + c];
    }
    __syncthreads();

    // gate logits: 40 (a,e) dot products spread over 8 waves
    for (int p = wave; p < An * En; p += 8) {
        const int a = p >> 2, e = p & 3;
        float s = 0.0f;
        for (int c = lane; c < Cn; c += 32)
            s = fmaf(attr[a * Cn + c], gate_w[e * Cn + c], s);
        s = wred(s);
        if (lane == 0) g[p] = s + gate_b[e];
    }
    // router bottleneck: t[a] = attr[a,:].r1_w + r1_b ; hval = gelu(t)
    for (int p = wave; p < An; p += 8) {
        float s = 0.0f;
        for (int c = lane; c < Cn; c += 32)
            s = fmaf(attr[p * Cn + c], r1_w[c], s);
        s = wred(s);
        if (lane == 0) {
            const float t = s + r1_b[0];
            hval[p] = 0.5f * t *
                (1.0f + tanhf(0.7978845608f * (t + 0.044715f * t * t * t)));
        }
    }
    if (wave == 0) {  // sums of r2_w, r2_b (means used below)
        float s1 = 0.0f, s2 = 0.0f;
        for (int c = lane; c < Cn; c += 32) { s1 += r2_w[c]; s2 += r2_b[c]; }
        s1 = wred(s1); s2 = wred(s2);
        if (lane == 0) { r2stats[0] = s1; r2stats[1] = s2; }
    }
    __syncthreads();

    if (tid == 0) {
        const float mr2w = r2stats[0] * (1.0f / (float)Cn);
        const float mr2b = r2stats[1] * (1.0f / (float)Cn);
        float sc10[An];
        for (int a = 0; a < An; ++a) sc10[a] = hval[a] * mr2w + mr2b;

        // per-attribute top-3-of-4 softmax over gate logits
        for (int a = 0; a < An; ++a) {
            float ga[En], t[En];
            for (int e = 0; e < En; ++e) { ga[e] = g[a * En + e]; t[e] = ga[e]; }
            for (int i = 0; i < En - 1; ++i)        // descending sort (tiny)
                for (int j = 0; j < En - 1 - i; ++j)
                    if (t[j] < t[j + 1]) { float tmp = t[j]; t[j] = t[j + 1]; t[j + 1] = tmp; }
            const float thr = t[MOE_TOPK - 1];
            const float mx = t[0];
            float s = 0.0f, we[En];
            for (int e = 0; e < En; ++e) {
                we[e] = (ga[e] >= thr) ? __expf(ga[e] - mx) : 0.0f;
                s += we[e];
            }
            for (int e = 0; e < En; ++e) w[a * En + e] = we[e] / s;
        }
        // top-7 attributes + softmax weights
        bool used[An] = {};
        float ts[TOPK_ATTR];
        for (int k = 0; k < TOPK_ATTR; ++k) {
            int bi = -1; float bv = -INFINITY;
            for (int a = 0; a < An; ++a)
                if (!used[a] && sc10[a] > bv) { bv = sc10[a]; bi = a; }
            used[bi] = true; ti[k] = bi; ts[k] = bv;
        }
        const float mx = ts[0];
        float s = 0.0f;
        for (int k = 0; k < TOPK_ATTR; ++k) { wts[k] = __expf(ts[k] - mx); s += wts[k]; }
        for (int k = 0; k < TOPK_ATTR; ++k) wts[k] /= s;
        for (int e = 0; e < En; ++e) {
            float c = 0.0f;
            for (int k = 0; k < TOPK_ATTR; ++k) c += wts[k] * w[ti[k] * En + e];
            ce[e] = c;
        }
    }
    __syncthreads();

    // z[b,e,c] and bias-init of output
#pragma unroll
    for (int cc = 0; cc < 3; ++cc) {
        const int c = tid + cc * 256;
        float sel[TOPK_ATTR];
#pragma unroll
        for (int k = 0; k < TOPK_ATTR; ++k)
            sel[k] = wts[k] * attr[ti[k] * Cn + c];
#pragma unroll
        for (int e = 0; e < En; ++e) {
            float zv = 0.0f;
#pragma unroll
            for (int k = 0; k < TOPK_ATTR; ++k)
                zv = fmaf(sel[k], w[ti[k] * En + e], zv);
            z[((size_t)b * En + e) * Cn + c] = zv;
        }
        float oi = 0.0f;
#pragma unroll
        for (int e = 0; e < En; ++e)
            oi = fmaf(ce[e], exp_b[e * Cn + c], oi);
        out_init[(size_t)b * Cn + c] = oi;
    }
}

// ---------------------------------------------------------------------------
// Host launcher
// ---------------------------------------------------------------------------
extern "C" void kernel_launch(void* const* d_in, const int* in_sizes, int n_in,
                              void* d_out, int out_size, void* d_ws, size_t ws_size,
                              hipStream_t stream) {
    const float* text_cls      = (const float*)d_in[0];   // [B,1,C]
    const float* visual_cls    = (const float*)d_in[1];   // [B,C]
    const float* visual_patchs = (const float*)d_in[2];   // [B,N,C] 192 MB
    const float* prompt        = (const float*)d_in[3];   // [1,A,C]
    const float* Wq = (const float*)d_in[4];  const float* bq = (const float*)d_in[5];
    const float* Wk = (const float*)d_in[6];  const float* bk = (const float*)d_in[7];
    const float* Wv = (const float*)d_in[8];  const float* bv = (const float*)d_in[9];
    const float* Wo = (const float*)d_in[10]; const float* bo = (const float*)d_in[11];
    const float* gate_w = (const float*)d_in[12]; const float* gate_b = (const float*)d_in[13];
    const float* exp_w  = (const float*)d_in[14]; const float* exp_b  = (const float*)d_in[15];
    const float* r1_w   = (const float*)d_in[16]; const float* r1_b   = (const float*)d_in[17];
    const float* r2_w   = (const float*)d_in[18]; const float* r2_b   = (const float*)d_in[19];
    float* out = (float*)d_out;

    // workspace carve (floats): ~19.3 MB total
    float* ws   = (float*)d_ws;
    float* q    = ws;                          // B*C
    float* q2   = q    + (size_t)Bn * Cn;      // B*H*C
    float* att  = q2   + (size_t)Bn * Hn * Cn; // B*H*N
    float* xbar = att  + (size_t)Bn * Hn * Nn; // B*H*C
    float* oatt = xbar + (size_t)Bn * Hn * Cn; // B*C
    float* mo   = oatt + (size_t)Bn * Cn;      // B*C
    float* z    = mo   + (size_t)Bn * Cn;      // B*E*C

    // 1) q = text_cls @ Wq^T + bq              [256,768]x[768,768]
    wmma_gemm_bt<false><<<dim3(Bn / 32, Cn / 32), 32, 0, stream>>>(
        text_cls, Cn, Wq, Cn, bq, q, Cn, Cn);

    // 2) fold q through Wk per head -> q2 [B,H,C]
    qhead_kernel<<<Bn, 256, 0, stream>>>(q, Wk, q2);

    // 3) streaming pass 1: scores + softmax -> att [B,H,N]
    scores_softmax_kernel<<<Bn, 256, 0, stream>>>(q2, visual_patchs, q, bk, att);

    // 4) streaming pass 2 (TDM double-buffered): att-weighted patch sum -> xbar
    xbar_kernel<<<Bn, 256, 0, stream>>>(att, visual_patchs, xbar);

    // 5) o[:, h*96:(h+1)*96] = xbar[:,h,:] @ Wv_h^T + bv_h   (8 WMMA GEMMs)
    for (int h = 0; h < Hn; ++h)
        wmma_gemm_bt<false><<<dim3(Bn / 32, Dn / 32), 32, 0, stream>>>(
            xbar + (size_t)h * Cn, Hn * Cn,
            Wv + (size_t)h * Dn * Cn, Cn,
            bv + h * Dn,
            oatt + h * Dn, Cn, Cn);

    // 6) moe_in = o @ Wo^T + bo
    wmma_gemm_bt<false><<<dim3(Bn / 32, Cn / 32), 32, 0, stream>>>(
        oatt, Cn, Wo, Cn, bo, mo, Cn, Cn);

    // 7) fused gate/router/top-k; builds z and bias-init of out
    moe_router_kernel<<<Bn, 256, 0, stream>>>(
        mo, visual_cls, prompt, gate_w, gate_b,
        r1_w, r1_b, r2_w, r2_b, exp_b, z, out);

    // 8) out += z[:,e,:] @ exp_w[e]^T   (4 accumulating WMMA GEMMs)
    for (int e = 0; e < En; ++e)
        wmma_gemm_bt<true><<<dim3(Bn / 32, Cn / 32), 32, 0, stream>>>(
            z + (size_t)e * Cn, En * Cn,
            exp_w + (size_t)e * Cn * Cn, Cn,
            nullptr, out, Cn, Cn);
}
